// CLConv_21715354649741
// MI455X (gfx1250) — matrix operand order repaired and checked by reading, not compiled, not run on hardware
//
#include <hip/hip_runtime.h>
#include <math.h>

typedef __attribute__((ext_vector_type(16))) _Float16 v16h;
typedef __attribute__((ext_vector_type(8)))  float    v8f;

#define N_NODES 4096
#define K_NB    16
#define EDGES   (N_NODES * K_NB)
#define FEAT    32
#define HID     64
#define BN_EPS  1e-5f

// ---------------------------------------------------------------------------
// Kernel A: per-edge MLP (WMMA for the 64x64 layer) + attention alpha.
// One wave = one node = one 16-edge tile (rows of the A matrix).
// Writes valsV[e] (relu MLP output) and valsA[e] (|q.k|) to workspace.
// ---------------------------------------------------------------------------
__global__ __launch_bounds__(256) void clconv_edge_mlp(
    const float* __restrict__ coor,                         // (E,4)
    const float* __restrict__ Wq, const float* __restrict__ bq,
    const float* __restrict__ Wk, const float* __restrict__ bk,
    const float* __restrict__ W0, const float* __restrict__ b0,
    const float* __restrict__ W1, const float* __restrict__ b1,
    const float* __restrict__ bn_g, const float* __restrict__ bn_b,
    const float* __restrict__ bn_m, const float* __restrict__ bn_v,
    const float* __restrict__ W2, const float* __restrict__ b2,
    float* __restrict__ valsV, float* __restrict__ valsA)
{
    // Stage W1 (64x64) in LDS as f16, padded rows to dodge bank conflicts.
    __shared__ _Float16 W1h[HID][72];
    const int tid = threadIdx.x;
    for (int i = tid; i < HID * HID; i += 256) {
        W1h[i >> 6][i & 63] = (_Float16)W1[i];
    }
    __syncthreads();

    const int wave = tid >> 5;                 // 8 waves / block
    const int lane = tid & 31;
    const int node = blockIdx.x * 8 + wave;    // grid = N/8
    const int m    = lane & 15;                // edge-in-node (lanes mirror 0..15)
    const int e    = node * K_NB + m;

    // coor + tanh for this edge (duplicated across half-waves; L1-served)
    const float c0 = coor[e * 4 + 0], c1 = coor[e * 4 + 1];
    const float c2 = coor[e * 4 + 2], c3 = coor[e * 4 + 3];
    const float t0 = tanhf(c0), t1 = tanhf(c1), t2 = tanhf(c2), t3 = tanhf(c3);

    // ---- A fragments: h0 = tanh(coor) @ W0 + b0, built directly in the
    // ISA 16-bit A layout: lane<16 -> K in {0..7, 16..23}; lane>=16 -> {8..15, 24..31}
    const int kb = (lane < 16) ? 0 : 8;
    v16h a0, a1;
    #pragma unroll
    for (int idx = 0; idx < 16; ++idx) {
        const int j0 = (idx < 8) ? (kb + idx) : (kb + 8 + idx);   // K chunk 0
        const int j1 = j0 + 32;                                    // K chunk 1
        float h0a = b0[j0] + t0 * W0[j0] + t1 * W0[64 + j0]
                           + t2 * W0[128 + j0] + t3 * W0[192 + j0];
        float h0b = b0[j1] + t0 * W0[j1] + t1 * W0[64 + j1]
                           + t2 * W0[128 + j1] + t3 * W0[192 + j1];
        a0[idx] = (_Float16)h0a;
        a1[idx] = (_Float16)h0b;
    }

    // ---- WMMA: (16 edges x 64) @ (64 x 64) -> 4 N-tiles x 2 K-chunks
    v8f acc[4];
    #pragma unroll
    for (int jt = 0; jt < 4; ++jt) acc[jt] = (v8f){0.f,0.f,0.f,0.f,0.f,0.f,0.f,0.f};

    #pragma unroll
    for (int jt = 0; jt < 4; ++jt) {
        v16h bf0, bf1;                         // B layout: lane%16 = N, elements = K (same order as A)
        #pragma unroll
        for (int n2 = 0; n2 < 16; ++n2) {
            bf0[n2] = W1h[kb + ((n2 < 8) ? n2 : n2 + 8)][jt * 16 + (lane & 15)];
            bf1[n2] = W1h[32 + kb + ((n2 < 8) ? n2 : n2 + 8)][jt * 16 + (lane & 15)];
        }
        acc[jt] = __builtin_amdgcn_wmma_f32_16x16x32_f16(
            false, a0, false, bf0, (short)0, acc[jt], false, false);
        acc[jt] = __builtin_amdgcn_wmma_f32_16x16x32_f16(
            false, a1, false, bf1, (short)0, acc[jt], false, false);
    }

    // ---- epilogue: +b1 -> tanh -> BN(eval) -> tanh -> @W2
    // C layout: VGPR r holds edge (r + 8*(lane>=16)), feature j = 16*jt + (lane&15)
    const int jc = lane & 15;
    float ysum[8];
    #pragma unroll
    for (int r = 0; r < 8; ++r) ysum[r] = 0.f;

    #pragma unroll
    for (int jt = 0; jt < 4; ++jt) {
        const int j = jt * 16 + jc;
        const float rstd = rsqrtf(bn_v[j] + BN_EPS);
        const float g  = bn_g[j] * rstd;
        const float bb = bn_b[j] - bn_m[j] * g;
        const float w2 = W2[j];
        const float bias = b1[j];
        #pragma unroll
        for (int r = 0; r < 8; ++r) {
            float z = acc[jt][r] + bias;
            z = tanhf(z);
            z = z * g + bb;
            ysum[r] += tanhf(z) * w2;
        }
    }
    // reduce over the 16 features held per half-wave (xor offsets stay in-half)
    #pragma unroll
    for (int off = 1; off < 16; off <<= 1) {
        #pragma unroll
        for (int r = 0; r < 8; ++r)
            ysum[r] += __shfl_xor(ysum[r], off, 32);
    }

    // ---- attention alpha for edge m: sphere = c[:2] + c[2:], q from edge 0
    const float s0 = c0 + c2, s1 = c1 + c3;
    const float qa = __shfl(s0, 0, 32);        // node-center sphere (edge 0)
    const float qb = __shfl(s1, 0, 32);
    float dot = 0.f;
    #pragma unroll
    for (int d = 0; d < 8; ++d) {
        const float q = bq[d] + qa * Wq[d] + qb * Wq[8 + d];
        const float k = bk[d] + s0 * Wk[d] + s1 * Wk[8 + d];
        dot += q * k;
    }
    const float alpha = fabsf(dot);

    // lane (jc<8, half h) finalizes edge mrow = jc + 8*h; fetch its alpha by shuffle
    const int mrow = jc + ((lane < 16) ? 0 : 8);
    const float alpha_w = __shfl(alpha, mrow, 32);   // lanes 0..15 own alpha[m]
    if (jc < 8) {
        const float v = fmaxf(ysum[jc] + b2[0], 0.f);
        const int ee = node * K_NB + mrow;
        valsV[ee] = v;
        valsA[ee] = alpha_w;
    }
}

// ---------------------------------------------------------------------------
// Kernel B: duplicate-col merge, w = (sumV)*exp(-(sumA)*geo)*ratio, then
// out[n, 32p+f] = sum_j w_j^(p+1) * x[col_j, f].  One wave per node.
// ---------------------------------------------------------------------------
__global__ __launch_bounds__(256) void clconv_combine_spmm(
    const float* __restrict__ x,
    const int*   __restrict__ sparse_idx,      // (2,E); col = +EDGES
    const float* __restrict__ geodesic,
    const float* __restrict__ angle_ratio,
    const float* __restrict__ valsV,
    const float* __restrict__ valsA,
    float* __restrict__ out)
{
    __shared__ float sw[8][3][16];
    __shared__ int   scol[8][16];

    const int tid  = threadIdx.x;
    const int wave = tid >> 5;
    const int lane = tid & 31;
    const int node = blockIdx.x * 8 + wave;
    const int m    = lane & 15;
    const int e    = node * K_NB + m;

    const int*   colp = sparse_idx + EDGES;
    const int    colm = colp[e];
    const float  vm   = valsV[e];
    const float  am   = valsA[e];

    // group duplicates among the node's 16 neighbors (reference scatter-adds
    // into the dense matrix BEFORE the elementwise power)
    float sumV = 0.f, sumA = 0.f;
    int leader = 16;
    #pragma unroll
    for (int j = 0; j < 16; ++j) {
        const int   cj = __shfl(colm, j, 32);
        const float vj = __shfl(vm,   j, 32);
        const float aj = __shfl(am,   j, 32);
        if (cj == colm) {
            sumV += vj; sumA += aj;
            if (j < leader) leader = j;
        }
    }

    float w = 0.f;
    if (leader == m) {   // first occurrence of this col carries the value
        const float g = geodesic[(size_t)node * N_NODES + colm];
        w = sumV * expf(-sumA * g) * angle_ratio[0];
    }
    if (lane < 16) {
        sw[wave][0][m] = w;
        sw[wave][1][m] = w * w;
        sw[wave][2][m] = w * w * w;
        scol[wave][m]  = colm;
    }
    __syncthreads();

    // lane = feature f (F=32); gather x rows, accumulate 3 views
    float acc0 = 0.f, acc1 = 0.f, acc2 = 0.f;
    #pragma unroll
    for (int mm = 0; mm < 16; ++mm) {
        const int   cc = scol[wave][mm];
        const float xv = x[(size_t)cc * FEAT + lane];
        acc0 += sw[wave][0][mm] * xv;
        acc1 += sw[wave][1][mm] * xv;
        acc2 += sw[wave][2][mm] * xv;
    }
    float* op = out + (size_t)node * (3 * FEAT);
    op[lane]            = acc0;
    op[FEAT + lane]     = acc1;
    op[2 * FEAT + lane] = acc2;
}

// ---------------------------------------------------------------------------
extern "C" void kernel_launch(void* const* d_in, const int* in_sizes, int n_in,
                              void* d_out, int out_size, void* d_ws, size_t ws_size,
                              hipStream_t stream) {
    const float* x     = (const float*)d_in[0];
    const float* coor  = (const float*)d_in[1];
    const int*   sidx  = (const int*)  d_in[2];
    const float* geo   = (const float*)d_in[3];
    const float* ratio = (const float*)d_in[4];
    const float* Wq    = (const float*)d_in[5];
    const float* bq    = (const float*)d_in[6];
    const float* Wk    = (const float*)d_in[7];
    const float* bk    = (const float*)d_in[8];
    const float* W0    = (const float*)d_in[9];
    const float* b0    = (const float*)d_in[10];
    const float* W1    = (const float*)d_in[11];
    const float* b1    = (const float*)d_in[12];
    const float* bn_g  = (const float*)d_in[13];
    const float* bn_b  = (const float*)d_in[14];
    const float* bn_m  = (const float*)d_in[15];
    const float* bn_v  = (const float*)d_in[16];
    const float* W2    = (const float*)d_in[17];
    const float* b2    = (const float*)d_in[18];
    // d_in[19] = max_view (fixed at 3 in the reference shapes)

    float* valsV = (float*)d_ws;           // E floats
    float* valsA = valsV + EDGES;          // E floats  (total 512 KB scratch)
    float* outp  = (float*)d_out;

    dim3 grid(N_NODES / 8), block(256);
    clconv_edge_mlp<<<grid, block, 0, stream>>>(
        coor, Wq, bq, Wk, bk, W0, b0, W1, b1,
        bn_g, bn_b, bn_m, bn_v, W2, b2, valsV, valsA);
    clconv_combine_spmm<<<grid, block, 0, stream>>>(
        x, sidx, geo, ratio, valsV, valsA, outp);
}
